// TFLlamaAttention_30880814858607
// MI455X (gfx1250) — matrix operand order repaired and checked by reading, not compiled, not run on hardware
//
#include <hip/hip_runtime.h>

// Problem constants (match reference)
#define DIMD   1024
#define BATCH  2
#define SEQ    2048
#define NHEAD  16
#define HDIM   64
#define BSROWS (BATCH * SEQ)        // 4096
#define KCH    (DIMD / 32)          // 32 k-chunks in projections
#define NEGBIG (-3.0e38f)

typedef __attribute__((ext_vector_type(16))) _Float16 v16h;
typedef __attribute__((ext_vector_type(8)))  float    v8f;

// ---------------------------------------------------------------------------
// Fragment helpers
// ---------------------------------------------------------------------------
// Packed fragment tile: 512 halves, lane l owns halves [l*16, l*16+16)
__device__ __forceinline__ v16h load_frag_pk(const _Float16* P, int tile, int lane) {
    return *reinterpret_cast<const v16h*>(P + (size_t)tile * 512 + lane * 16);
}

// Async 16B global -> LDS copy (per lane), tracked by ASYNCcnt.
// LDS byte address = low 32 bits of the generic pointer to __shared__.
__device__ __forceinline__ void async_copy_16B(const _Float16* src, _Float16* ldsDst) {
    unsigned dst = (unsigned)(uintptr_t)ldsDst;
    unsigned long long a = (unsigned long long)(uintptr_t)src;
    asm volatile("global_load_async_to_lds_b128 %0, %1, off"
                 :: "v"(dst), "v"(a) : "memory");
}
__device__ __forceinline__ void wait_async0() {
    asm volatile("s_wait_asynccnt 0" ::: "memory");
}

// ---------------------------------------------------------------------------
// 1) Pack f32 weight [K=D rows, N=D cols] into f16 B-fragment tiles.
// B-fragment (32x16): lane l -> col n0+(l&15); halves j=0..15 -> row k0+(l>>4)*16+j
// Tile index = kChunk * (D/16) + nTile; 512 halves per tile, lane*16 contiguous.
// ---------------------------------------------------------------------------
__global__ void pack_weight_kernel(const float* __restrict__ W, _Float16* __restrict__ P) {
    const int t    = blockIdx.x * blockDim.x + threadIdx.x;  // one thread per lane-slot
    const int lane = t & 31;
    const int tile = t >> 5;
    const int ntile = tile & 63;            // D/16 = 64 column tiles
    const int kch   = tile >> 6;            // D/32 = 32 k-chunks
    const int n  = ntile * 16 + (lane & 15);
    const int kb = kch * 32 + (lane >> 4) * 16;
    _Float16* dst = P + (size_t)tile * 512 + lane * 16;
#pragma unroll
    for (int j = 0; j < 16; ++j)
        dst[j] = (_Float16)W[(size_t)(kb + j) * DIMD + n];
}

// ---------------------------------------------------------------------------
// 2) C(f32) = A(f32) * Bpk(f16 packed), K = DIMD.
//    Register blocking: each wave computes 4 adjacent 16x16 N-tiles, reusing
//    one A-fragment for 4 WMMAs. Pointer-stepped addressing: the 4 B tiles sit
//    at constant offsets (0/1/2/3 KB) from one base that advances once per
//    k-chunk, so the backend can clause the b128 loads and overlap with WMMA.
// ---------------------------------------------------------------------------
__global__ void __launch_bounds__(128) gemm_wmma_kernel(const float* __restrict__ A,
                                                        const _Float16* __restrict__ Bpk,
                                                        float* __restrict__ C,
                                                        int M, int N) {
    const int wave = threadIdx.x >> 5;
    const int lane = threadIdx.x & 31;
    const int l16  = lane & 15;
    const int hi   = lane >> 4;
    const int m0   = blockIdx.x * 16;
    const int ntBase = (blockIdx.y * 4 + wave) * 4;   // 4 N-tiles per wave
    const int nTiles = N >> 4;
    const size_t bStep = (size_t)nTiles * 512;        // halves per k-chunk of B

    const float*    arow = A   + (size_t)(m0 + l16) * DIMD + hi * 16;  // A row, this lane
    const _Float16* bp   = Bpk + (size_t)ntBase * 512 + lane * 16;    // first B tile

    v8f acc0 = {}, acc1 = {}, acc2 = {}, acc3 = {};
#pragma unroll 2
    for (int kc = 0; kc < KCH; ++kc) {
        v16h a;
#pragma unroll
        for (int j = 0; j < 8; ++j) {
            a[j]     = (_Float16)arow[j];        // K = k0 + hi*16 + j
            a[j + 8] = (_Float16)arow[8 + j];    // K = k0 + hi*16 + 8 + j
        }
        __builtin_prefetch(bp + bStep, 0, 1);    // speculative; harmless past end
        v16h b0 = *reinterpret_cast<const v16h*>(bp);
        v16h b1 = *reinterpret_cast<const v16h*>(bp + 512);
        v16h b2 = *reinterpret_cast<const v16h*>(bp + 1024);
        v16h b3 = *reinterpret_cast<const v16h*>(bp + 1536);
        acc0 = __builtin_amdgcn_wmma_f32_16x16x32_f16(false, a, false, b0, (short)0, acc0, false, false);
        acc1 = __builtin_amdgcn_wmma_f32_16x16x32_f16(false, a, false, b1, (short)0, acc1, false, false);
        acc2 = __builtin_amdgcn_wmma_f32_16x16x32_f16(false, a, false, b2, (short)0, acc2, false, false);
        acc3 = __builtin_amdgcn_wmma_f32_16x16x32_f16(false, a, false, b3, (short)0, acc3, false, false);
        arow += 32;
        bp   += bStep;
    }

    float* crow = C + (size_t)(m0 + hi * 8) * N + ntBase * 16 + l16;
#pragma unroll
    for (int r = 0; r < 8; ++r) {
        crow[0]  = acc0[r];
        crow[16] = acc1[r];
        crow[32] = acc2[r];
        crow[48] = acc3[r];
        crow += N;
    }
}

// ---------------------------------------------------------------------------
// 3) RoPE + pack Q (A-layout), K^T (B-layout), V (B-layout) as f16 per (b,h).
//    mode = blockIdx.y: 0=Q, 1=K, 2=V. One thread per packed lane-slot.
//    Per (b,h): 256 tiles * 512 halves = 131072 halves.
// ---------------------------------------------------------------------------
__device__ __forceinline__ float rope_apply(const float* __restrict__ head, int pos, int hd) {
    float x  = head[hd];
    float xr = (hd < 32) ? -head[hd + 32] : head[hd - 32];
    // inv_freq[j] = 10000^(-j/32) = exp(-j * ln(1e4)/32), j = hd & 31
    float ang = (float)pos * __expf(-(float)(hd & 31) * 0.2878231366242557f);
    return x * __cosf(ang) + xr * __sinf(ang);
}

__global__ void rope_pack_kernel(const float* __restrict__ Qf, const float* __restrict__ Kf,
                                 const float* __restrict__ Vf, const int* __restrict__ pid,
                                 _Float16* __restrict__ Qpk, _Float16* __restrict__ Kpk,
                                 _Float16* __restrict__ Vpk) {
    const int mode = blockIdx.y;
    const int t    = blockIdx.x * blockDim.x + threadIdx.x;
    const int lane = t & 31;
    const int l16  = lane & 15;
    const int hi   = (lane >> 4) & 1;
    const int tile = (t >> 5) & 255;   // 256 tiles per (b,h)
    const int bh   = t >> 13;          // 8192 slots per (b,h)
    const int b = bh / NHEAD, h = bh % NHEAD;

    if (mode == 0) {
        // Q: A-fragment layout. tile = qTile*2 + hdChunk
        const int qTile = tile >> 1, hdc = tile & 1;
        const int s = qTile * 16 + l16;
        const float* head = Qf + (size_t)(b * SEQ + s) * DIMD + h * HDIM;
        const int pos = pid[b * SEQ + s];
        _Float16* dst = Qpk + (size_t)bh * 131072 + (size_t)tile * 512 + lane * 16;
#pragma unroll
        for (int j = 0; j < 16; ++j) {
            const int hd = hdc * 32 + hi * 16 + j;
            dst[j] = (_Float16)rope_apply(head, pos, hd);
        }
    } else if (mode == 1) {
        // K^T: B-fragment layout, rows = hd, cols = kv. tile = kch*128 + nTile
        const int kch = tile >> 7, ntv = tile & 127;
        const int s = ntv * 16 + l16;
        const float* head = Kf + (size_t)(b * SEQ + s) * DIMD + h * HDIM;
        const int pos = pid[b * SEQ + s];
        _Float16* dst = Kpk + (size_t)bh * 131072 + (size_t)tile * 512 + lane * 16;
#pragma unroll
        for (int j = 0; j < 16; ++j) {
            const int hd = kch * 32 + hi * 16 + j;
            dst[j] = (_Float16)rope_apply(head, pos, hd);
        }
    } else {
        // V: B-fragment layout, rows = kv, cols = hd. tile = kch*4 + nTile
        const int kch = tile >> 2, nt = tile & 3;
        const int hd = nt * 16 + l16;
        const float* col = Vf + (size_t)b * SEQ * DIMD + h * HDIM + hd;
        _Float16* dst = Vpk + (size_t)bh * 131072 + (size_t)tile * 512 + lane * 16;
#pragma unroll
        for (int j = 0; j < 16; ++j) {
            const int kv = kch * 32 + hi * 16 + j;
            dst[j] = (_Float16)col[(size_t)kv * DIMD];
        }
    }
}

// ---------------------------------------------------------------------------
// 4) Flash attention per (b,h). Block = 4 waves; wave owns 16 q rows of a
//    64-row q block. kv processed in 32-key chunks with online softmax.
//    K/V chunks (shared by all 4 waves) are double-buffered in LDS via
//    global_load_async_to_lds_b128, overlapped with the previous chunk's math.
// ---------------------------------------------------------------------------
__device__ __forceinline__ void issue_kv_chunk(const _Float16* __restrict__ Kb,
                                               const _Float16* __restrict__ Vb,
                                               int kc, _Float16* kdst, _Float16* vdst,
                                               int t /*0..127*/) {
    // K tiles for chunk kc: (kch0,nt0)=kc*2, (kch0,nt1)=kc*2+1,
    //                       (kch1,nt0)=128+kc*2, (kch1,nt1)=128+kc*2+1
    const int ktiles[4] = {kc * 2, kc * 2 + 1, 128 + kc * 2, 128 + kc * 2 + 1};
    // 512 x 16B segments total: segs 0..255 -> K (4 tiles x 64), 256..511 -> V
#pragma unroll
    for (int q = 0; q < 4; ++q) {
        const int seg = t + 128 * q;
        if (seg < 256) {
            const int tl = seg >> 6, w = seg & 63;
            async_copy_16B(Kb + (size_t)ktiles[tl] * 512 + w * 8, kdst + tl * 512 + w * 8);
        } else {
            const int s2 = seg - 256;
            const int tl = s2 >> 6, w = s2 & 63;
            async_copy_16B(Vb + (size_t)(kc * 4 + tl) * 512 + w * 8, vdst + tl * 512 + w * 8);
        }
    }
}

__device__ __forceinline__ v16h lds_frag(const _Float16* base, int tile, int lane) {
    return *reinterpret_cast<const v16h*>(base + tile * 512 + lane * 16);
}

__global__ void __launch_bounds__(128) attn_kernel(const _Float16* __restrict__ Qpk,
                                                   const _Float16* __restrict__ Kpk,
                                                   const _Float16* __restrict__ Vpk,
                                                   float* __restrict__ O) {
    __shared__ __align__(32) _Float16 pTile[4][512];     // 16x32 f16 P tile per wave
    __shared__ __align__(32) _Float16 kbuf[2][4 * 512];  // double-buffered K chunk (4 tiles)
    __shared__ __align__(32) _Float16 vbuf[2][4 * 512];  // double-buffered V chunk (4 tiles)

    const int tid  = threadIdx.x;
    const int wave = tid >> 5;
    const int lane = tid & 31;
    const int l16  = lane & 15;
    const int hi   = lane >> 4;
    const int bh   = blockIdx.x;
    const int b = bh / NHEAD, h = bh % NHEAD;
    const int qBlock = blockIdx.y;
    const int m0 = qBlock * 64 + wave * 16;

    const _Float16* Qb = Qpk + (size_t)bh * 131072;
    const _Float16* Kb = Kpk + (size_t)bh * 131072;
    const _Float16* Vb = Vpk + (size_t)bh * 131072;

    const int qTile = m0 >> 4;
    const v16h qa0 = load_frag_pk(Qb, qTile * 2 + 0, lane);   // hd 0..31
    const v16h qa1 = load_frag_pk(Qb, qTile * 2 + 1, lane);   // hd 32..63

    v8f acc0 = {}, acc1 = {}, acc2 = {}, acc3 = {};
    float mrow[8], lrow[8];
#pragma unroll
    for (int r = 0; r < 8; ++r) { mrow[r] = NEGBIG; lrow[r] = 0.0f; }

    const int kcEnd = qBlock * 2 + 1;   // uniform over the block -> barriers legal
    int buf = 0;
    issue_kv_chunk(Kb, Vb, 0, kbuf[0], vbuf[0], tid);

    for (int kc = 0; kc <= kcEnd; ++kc) {
        const int kvb = kc * 32;

        wait_async0();        // this wave's async stores to LDS have landed
        __syncthreads();      // ... and every other wave's too
        if (kc < kcEnd)       // overlap next chunk's stream with this chunk's math
            issue_kv_chunk(Kb, Vb, kc + 1, kbuf[buf ^ 1], vbuf[buf ^ 1], tid);

        const _Float16* kcur = kbuf[buf];
        const _Float16* vcur = vbuf[buf];

        // ---- S = Q K^T (16 q rows x 32 keys), accumulate over hd chunks ----
        v8f s0 = {}, s1 = {};
        {
            v16h kb00 = lds_frag(kcur, 0, lane);   // (kch0, nt0)
            v16h kb01 = lds_frag(kcur, 1, lane);   // (kch0, nt1)
            v16h kb10 = lds_frag(kcur, 2, lane);   // (kch1, nt0)
            v16h kb11 = lds_frag(kcur, 3, lane);   // (kch1, nt1)
            s0 = __builtin_amdgcn_wmma_f32_16x16x32_f16(false, qa0, false, kb00, (short)0, s0, false, false);
            s0 = __builtin_amdgcn_wmma_f32_16x16x32_f16(false, qa1, false, kb10, (short)0, s0, false, false);
            s1 = __builtin_amdgcn_wmma_f32_16x16x32_f16(false, qa0, false, kb01, (short)0, s1, false, false);
            s1 = __builtin_amdgcn_wmma_f32_16x16x32_f16(false, qa1, false, kb11, (short)0, s1, false, false);
        }

        // ---- scale, causal mask, online softmax (rows striped per C-layout) ----
        float alphaR[8];
#pragma unroll
        for (int r = 0; r < 8; ++r) {
            const int qg = m0 + r + hi * 8;          // global q row for this lane/VGPR
            float x0 = s0[r] * 0.125f;               // 1/sqrt(64)
            float x1 = s1[r] * 0.125f;
            if (kvb + l16 > qg)      x0 = NEGBIG;
            if (kvb + 16 + l16 > qg) x1 = NEGBIG;

            float cm = fmaxf(x0, x1);
#pragma unroll
            for (int off = 1; off < 16; off <<= 1)
                cm = fmaxf(cm, __shfl_xor(cm, off, 16));
            const float mnew = fmaxf(mrow[r], cm);

            const float p0 = __expf(x0 - mnew);
            const float p1 = __expf(x1 - mnew);
            float psum = p0 + p1;
#pragma unroll
            for (int off = 1; off < 16; off <<= 1)
                psum += __shfl_xor(psum, off, 16);

            const float alpha = __expf(mrow[r] - mnew);
            lrow[r]  = lrow[r] * alpha + psum;
            mrow[r]  = mnew;
            alphaR[r] = alpha;

            pTile[wave][(r + hi * 8) * 32 + l16]      = (_Float16)p0;
            pTile[wave][(r + hi * 8) * 32 + 16 + l16] = (_Float16)p1;
        }
        __syncthreads();

        // ---- C-layout -> A-layout P fragment via LDS ----
        v16h pa;
        {
            const _Float16* pr = &pTile[wave][l16 * 32];
#pragma unroll
            for (int j = 0; j < 8; ++j) {
                pa[j]     = pr[hi * 16 + j];
                pa[j + 8] = pr[hi * 16 + 8 + j];
            }
        }

        // ---- rescale accumulators, then O += P V ----
#pragma unroll
        for (int r = 0; r < 8; ++r) {
            acc0[r] *= alphaR[r]; acc1[r] *= alphaR[r];
            acc2[r] *= alphaR[r]; acc3[r] *= alphaR[r];
        }
        {
            v16h vb0 = lds_frag(vcur, 0, lane);
            v16h vb1 = lds_frag(vcur, 1, lane);
            v16h vb2 = lds_frag(vcur, 2, lane);
            v16h vb3 = lds_frag(vcur, 3, lane);
            acc0 = __builtin_amdgcn_wmma_f32_16x16x32_f16(false, pa, false, vb0, (short)0, acc0, false, false);
            acc1 = __builtin_amdgcn_wmma_f32_16x16x32_f16(false, pa, false, vb1, (short)0, acc1, false, false);
            acc2 = __builtin_amdgcn_wmma_f32_16x16x32_f16(false, pa, false, vb2, (short)0, acc2, false, false);
            acc3 = __builtin_amdgcn_wmma_f32_16x16x32_f16(false, pa, false, vb3, (short)0, acc3, false, false);
        }
        __syncthreads();   // protects pTile reuse AND kv buffer recycling
        buf ^= 1;
    }

    // ---- normalize and store f32 [BS, D] (col = h*64 + hd) ----
#pragma unroll
    for (int r = 0; r < 8; ++r) {
        const float inv = 1.0f / lrow[r];
        float* op = O + (size_t)(b * SEQ + m0 + r + hi * 8) * DIMD + h * HDIM + l16;
        op[0]  = acc0[r] * inv;
        op[16] = acc1[r] * inv;
        op[32] = acc2[r] * inv;
        op[48] = acc3[r] * inv;
    }
}

// ---------------------------------------------------------------------------
// Launch
// ---------------------------------------------------------------------------
extern "C" void kernel_launch(void* const* d_in, const int* in_sizes, int n_in,
                              void* d_out, int out_size, void* d_ws, size_t ws_size,
                              hipStream_t stream) {
    (void)in_sizes; (void)n_in; (void)out_size; (void)ws_size;
    const float* hid = (const float*)d_in[0];
    // d_in[1] = attention_mask (causal; recomputed analytically on device)
    const float* wq  = (const float*)d_in[2];
    const float* wk  = (const float*)d_in[3];
    const float* wv  = (const float*)d_in[4];
    const float* wo  = (const float*)d_in[5];
    const int*   pid = (const int*)d_in[6];

    char* ws = (char*)d_ws;
    size_t off = 0;
    const size_t W_PK_BYTES  = (size_t)DIMD * DIMD * 2;                 // 2 MB each
    const size_t F32_BYTES   = (size_t)BSROWS * DIMD * 4;               // 16 MB each
    const size_t PK_BYTES    = (size_t)BATCH * NHEAD * SEQ * HDIM * 2;  // 8 MB each

    _Float16* wq_pk = (_Float16*)(ws + off); off += W_PK_BYTES;
    _Float16* wk_pk = (_Float16*)(ws + off); off += W_PK_BYTES;
    _Float16* wv_pk = (_Float16*)(ws + off); off += W_PK_BYTES;
    _Float16* wo_pk = (_Float16*)(ws + off); off += W_PK_BYTES;
    float* qf = (float*)(ws + off); off += F32_BYTES;
    float* kf = (float*)(ws + off); off += F32_BYTES;
    float* vf = (float*)(ws + off); off += F32_BYTES;
    _Float16* qpk = (_Float16*)(ws + off); off += PK_BYTES;
    _Float16* kpk = (_Float16*)(ws + off); off += PK_BYTES;
    _Float16* vpk = (_Float16*)(ws + off); off += PK_BYTES;
    float* attn = (float*)(ws + off); off += F32_BYTES;

    // 1) pack weights into B-fragment layout (65536 lane-slots each)
    pack_weight_kernel<<<256, 256, 0, stream>>>(wq, wq_pk);
    pack_weight_kernel<<<256, 256, 0, stream>>>(wk, wk_pk);
    pack_weight_kernel<<<256, 256, 0, stream>>>(wv, wv_pk);
    pack_weight_kernel<<<256, 256, 0, stream>>>(wo, wo_pk);

    // 2) QKV projections (WMMA, 4 N-tiles per wave)
    dim3 gGemm(BSROWS / 16, (DIMD / 16) / 16);   // (256, 4)
    gemm_wmma_kernel<<<gGemm, 128, 0, stream>>>(hid, wq_pk, qf, BSROWS, DIMD);
    gemm_wmma_kernel<<<gGemm, 128, 0, stream>>>(hid, wk_pk, kf, BSROWS, DIMD);
    gemm_wmma_kernel<<<gGemm, 128, 0, stream>>>(hid, wv_pk, vf, BSROWS, DIMD);

    // 3) RoPE + fragment packing (262144 lane-slots per tensor)
    rope_pack_kernel<<<dim3(1024, 3), 256, 0, stream>>>(qf, kf, vf, pid, qpk, kpk, vpk);

    // 4) fused causal flash attention (WMMA + async K/V streaming to LDS)
    attn_kernel<<<dim3(BATCH * NHEAD, SEQ / 64), 128, 0, stream>>>(qpk, kpk, vpk, attn);

    // 5) output projection (WMMA) -> d_out f32
    gemm_wmma_kernel<<<gGemm, 128, 0, stream>>>(attn, wo_pk, (float*)d_out, BSROWS, DIMD);
}
